// SelfAttention_25924422598785
// MI455X (gfx1250) — matrix operand order repaired and checked by reading, not compiled, not run on hardware
//
#include <hip/hip_runtime.h>

#define DIMN  1024
#define HNUM  16
#define DH    64
#define TLEN  4096
#define BATCH 4
#define MTOT  (BATCH * TLEN)   /* 16384 */
#define NQKV  (3 * DIMN)       /* 3072  */

typedef __bf16 bf16_t;
typedef __attribute__((ext_vector_type(16))) __bf16 bf16x16;
typedef __attribute__((ext_vector_type(8)))  __bf16 bf16x8;
typedef __attribute__((ext_vector_type(8)))  float  f32x8;

// ---------------- WMMA fragment helpers (wave32, 16x16x32 bf16) ----------------

// A fragment: 16x32 (MxK), row-major source with leading dim `ld` (elements).
// lanes 0-15: row=lane, K in {0..7, 16..23}; lanes 16-31: row=lane-16, K in {8..15, 24..31}.
__device__ __forceinline__ bf16x16 load_a_frag(const bf16_t* base, int ld) {
    const int lane = threadIdx.x & 31;
    const int r = lane & 15, hf = lane >> 4;
    const bf16_t* p = base + (size_t)r * ld + hf * 8;
    bf16x8 lo = *(const bf16x8*)(p);
    bf16x8 hi = *(const bf16x8*)(p + 16);
    return __builtin_shufflevector(lo, hi, 0,1,2,3,4,5,6,7,8,9,10,11,12,13,14,15);
}

// B fragment: 32x16 (KxN) from N-major storage Bt[N][K] with leading dim `ld`.
// lanes 0-15: n=lane, K in {0..7, 16..23}; lanes 16-31: n=lane-16, K in {8..15, 24..31}.
__device__ __forceinline__ bf16x16 load_bt_frag(const bf16_t* base, int ld) {
    const int lane = threadIdx.x & 31;
    const int n = lane & 15, hf = lane >> 4;
    const bf16_t* p = base + (size_t)n * ld + hf * 8;
    bf16x8 lo = *(const bf16x8*)(p);
    bf16x8 hi = *(const bf16x8*)(p + 16);
    return __builtin_shufflevector(lo, hi, 0,1,2,3,4,5,6,7,8,9,10,11,12,13,14,15);
}

__device__ __forceinline__ f32x8 wmma_bf16(bf16x16 a, bf16x16 b, f32x8 c) {
    return __builtin_amdgcn_wmma_f32_16x16x32_bf16(false, a, false, b, (short)0, c,
                                                   false, false);
}

// ---------------- CDNA5 async copy: global -> LDS (ASYNCcnt) ----------------

// Low 32 bits of a generic pointer into __shared__ = LDS byte address (flat-LDS rule).
__device__ __forceinline__ unsigned lds_addr(const void* p) {
    return (unsigned)(unsigned long long)(uintptr_t)p;
}

__device__ __forceinline__ void async_load_b128(unsigned lds_off, const void* gptr) {
    asm volatile("global_load_async_to_lds_b128 %0, %1, off"
                 :
                 : "v"(lds_off), "v"((unsigned long long)(uintptr_t)gptr)
                 : "memory");
}

__device__ __forceinline__ void wait_async0() {
    asm volatile("s_wait_asynccnt 0x0" ::: "memory");
}

// ---------------- prep kernels ----------------

__global__ void cast_f32_bf16_kernel(const float* __restrict__ in,
                                     bf16_t* __restrict__ out, int n) {
    int i = blockIdx.x * blockDim.x + threadIdx.x;
    if (i < n) out[i] = (bf16_t)in[i];
}

// wqkv_t[n][k] = W{q|k|v}[k][n mod 1024], n in [0,3072)
__global__ void pack_wqkv_kernel(const float* __restrict__ Wq,
                                 const float* __restrict__ Wk,
                                 const float* __restrict__ Wv,
                                 bf16_t* __restrict__ wt) {
    int i = blockIdx.x * blockDim.x + threadIdx.x;
    if (i >= NQKV * DIMN) return;
    int n = i >> 10, k = i & (DIMN - 1);
    float v;
    if (n < DIMN)            v = Wq[(size_t)k * DIMN + n];
    else if (n < 2 * DIMN)   v = Wk[(size_t)k * DIMN + (n - DIMN)];
    else                     v = Wv[(size_t)k * DIMN + (n - 2 * DIMN)];
    wt[i] = (bf16_t)v;
}

__global__ void pack_wo_kernel(const float* __restrict__ Wo, bf16_t* __restrict__ wt) {
    int i = blockIdx.x * blockDim.x + threadIdx.x;
    if (i >= DIMN * DIMN) return;
    int n = i >> 10, k = i & (DIMN - 1);
    wt[i] = (bf16_t)Wo[(size_t)k * DIMN + n];
}

// ---------------- fused QKV projection GEMM (async-LDS double-buffered) ----------------
// C[16384,3072] = x_bf[16384,1024] * wqkv_t. Block = 128x128 tile, 8 waves,
// wave = 32(M) x 64(N). K staged in 64-wide double-buffered LDS tiles via
// global_load_async_to_lds_b128; compute consumes via ds_load_b128.
__global__ void __launch_bounds__(256) qkv_gemm_kernel(const bf16_t* __restrict__ xbf,
                                                       const bf16_t* __restrict__ wt,
                                                       float* __restrict__ qf,
                                                       float* __restrict__ ktf,
                                                       bf16_t* __restrict__ vt) {
    __shared__ bf16_t tA[2][128 * 64];
    __shared__ bf16_t tB[2][128 * 64];

    const int tid = threadIdx.x;
    const int w = tid >> 5, lane = tid & 31;
    const int wm = w & 3, wn = w >> 2;
    const int m0 = blockIdx.y * 128;
    const int n0 = blockIdx.x * 128;
    const bf16_t* agbl = xbf + (size_t)m0 * DIMN;
    const bf16_t* bgbl = wt  + (size_t)n0 * DIMN;

    f32x8 acc[2][4] = {};

    auto stage = [&](int s) {
        const int b = s & 1;
        const int k0 = s * 64;
#pragma unroll
        for (int i = 0; i < 4; ++i) {
            int chunk = tid + i * 256;          // 0..1023 -> (row, 8-elem col group)
            int row = chunk >> 3;
            int kc  = (chunk & 7) * 8;
            async_load_b128(lds_addr(&tA[b][chunk * 8]),
                            agbl + (size_t)row * DIMN + k0 + kc);
            async_load_b128(lds_addr(&tB[b][chunk * 8]),
                            bgbl + (size_t)row * DIMN + k0 + kc);
        }
    };

    auto compute = [&](int b) {
        const bf16_t* As = &tA[b][(wm * 32) * 64];
        const bf16_t* Bs = &tB[b][(wn * 64) * 64];
#pragma unroll
        for (int kk = 0; kk < 64; kk += 32) {
            bf16x16 a0 = load_a_frag(As + kk, 64);
            bf16x16 a1 = load_a_frag(As + 16 * 64 + kk, 64);
#pragma unroll
            for (int j = 0; j < 4; ++j) {
                bf16x16 bb = load_bt_frag(Bs + (j * 16) * 64 + kk, 64);
                acc[0][j] = wmma_bf16(a0, bb, acc[0][j]);
                acc[1][j] = wmma_bf16(a1, bb, acc[1][j]);
            }
        }
    };

    stage(0);
    wait_async0();
    __syncthreads();
    const int KT = DIMN / 64;                   // 16 stages
    for (int s = 0; s < KT; ++s) {
        if (s + 1 < KT) stage(s + 1);           // DMA next tile while computing
        compute(s & 1);
        wait_async0();                          // our stage s+1 ops landed
        __syncthreads();                        // everyone's landed; buffer reuse safe
    }

    const int hf = lane >> 4, nn = lane & 15;
    const int region = blockIdx.x >> 3;         // 0:q 1:k 2:v (tile never crosses regions)
    const int wm0 = m0 + wm * 32, wn0 = n0 + wn * 64;
#pragma unroll
    for (int i = 0; i < 2; ++i)
#pragma unroll
        for (int j = 0; j < 4; ++j)
#pragma unroll
            for (int e = 0; e < 8; ++e) {
                int r = wm0 + i * 16 + e + 8 * hf;     // global row = b*T + t
                int c = wn0 + j * 16 + nn;             // global col in [0,3072)
                float val = acc[i][j][e];
                if (region == 0) {
                    qf[(size_t)r * DIMN + c] = val;
                } else if (region == 1) {
                    int c2 = c - DIMN, hh = c2 >> 6, d = c2 & 63;
                    int bb2 = r >> 12, t = r & (TLEN - 1);
                    ktf[(((size_t)(bb2 * HNUM + hh)) * DH + d) * TLEN + t] = val;
                } else {
                    int c2 = c - 2 * DIMN, hh = c2 >> 6, d = c2 & 63;
                    int bb2 = r >> 12, t = r & (TLEN - 1);
                    vt[(((size_t)(bb2 * HNUM + hh)) * DH + d) * TLEN + t] = (bf16_t)val;
                }
            }
}

// ---------------- softmax over feature dim for q (64 contiguous), * DH^-0.5 ----------------
__global__ void softmax_q_kernel(const float* __restrict__ qf, bf16_t* __restrict__ qbf) {
    int g = (blockIdx.x * blockDim.x + threadIdx.x) >> 5;  // wave group: row*16+head
    int lane = threadIdx.x & 31;
    int row = g >> 4, head = g & 15;
    const float* p = qf + (size_t)row * DIMN + head * DH + lane * 2;
    float v0 = p[0], v1 = p[1];
    float m = fmaxf(v0, v1);
    for (int off = 16; off; off >>= 1) m = fmaxf(m, __shfl_xor(m, off, 32));
    float e0 = __expf(v0 - m), e1 = __expf(v1 - m);
    float s = e0 + e1;
    for (int off = 16; off; off >>= 1) s += __shfl_xor(s, off, 32);
    float inv = 0.125f / s;   // DH^-0.5 = 1/8
    bf16_t* q = qbf + (size_t)row * DIMN + head * DH + lane * 2;
    q[0] = (bf16_t)(e0 * inv);
    q[1] = (bf16_t)(e1 * inv);
}

// ---------------- softmax over seq dim for k; k_t layout makes t contiguous ----------------
__global__ void softmax_k_kernel(const float* __restrict__ ktf, bf16_t* __restrict__ kbf) {
    __shared__ float red[256];
    const int rid = blockIdx.x;             // (b*16+h)*64+d, 4096 rows
    const float* p = ktf + (size_t)rid * TLEN;
    const int tid = threadIdx.x;
    float m = -1e30f;
    for (int i = tid; i < TLEN; i += 256) m = fmaxf(m, p[i]);
    red[tid] = m; __syncthreads();
    for (int s = 128; s; s >>= 1) { if (tid < s) red[tid] = fmaxf(red[tid], red[tid + s]); __syncthreads(); }
    m = red[0]; __syncthreads();
    float sum = 0.f;
    for (int i = tid; i < TLEN; i += 256) sum += __expf(p[i] - m);
    red[tid] = sum; __syncthreads();
    for (int s = 128; s; s >>= 1) { if (tid < s) red[tid] += red[tid + s]; __syncthreads(); }
    float inv = 1.0f / red[0];
    bf16_t* o = kbf + (size_t)rid * TLEN;
    for (int i = tid; i < TLEN; i += 256) o[i] = (bf16_t)(__expf(p[i] - m) * inv);
}

// ---------------- context[d][e] = sum_t k[t,d] v[t,e], per (b,h); store ctx^T bf16 ----------------
__global__ void __launch_bounds__(128) context_kernel(const bf16_t* __restrict__ kbf,
                                                      const bf16_t* __restrict__ vt,
                                                      bf16_t* __restrict__ ctxt) {
    const int bh = blockIdx.x;                 // 64
    const int w = threadIdx.x >> 5;            // 4 waves, wave owns d-strip
    const int lane = threadIdx.x & 31;
    const bf16_t* kbase = kbf + (size_t)bh * DH * TLEN + (size_t)(w * 16) * TLEN;
    const bf16_t* vbase = vt  + (size_t)bh * DH * TLEN;
    f32x8 acc[4] = {};
    for (int k0 = 0; k0 < TLEN; k0 += 32) {
        bf16x16 a = load_a_frag(kbase + k0, TLEN);
#pragma unroll
        for (int j = 0; j < 4; ++j) {
            bf16x16 b = load_bt_frag(vbase + (size_t)(j * 16) * TLEN + k0, TLEN);
            acc[j] = wmma_bf16(a, b, acc[j]);
        }
    }
    const int hf = lane >> 4, nn = lane & 15;
    bf16_t* out = ctxt + (size_t)bh * DH * DH;   // ctx^T: [e][d]
#pragma unroll
    for (int j = 0; j < 4; ++j)
#pragma unroll
        for (int e = 0; e < 8; ++e) {
            int d = w * 16 + e + 8 * hf;
            int col = j * 16 + nn;
            out[(size_t)col * DH + d] = (bf16_t)acc[j][e];
        }
}

// ---------------- attn[t][e] = q[t][:] * ctx[:,e], per (b,h), t-tiled ----------------
__global__ void __launch_bounds__(128) attn_kernel(const bf16_t* __restrict__ qbf,
                                                   const bf16_t* __restrict__ ctxt,
                                                   bf16_t* __restrict__ attnbf) {
    const int bh = blockIdx.y, b = bh >> 4, hd = bh & 15;
    const int w = threadIdx.x >> 5, lane = threadIdx.x & 31;
    const int t0 = blockIdx.x * 128 + w * 32;
    const bf16_t* abase = qbf + ((size_t)(b * TLEN + t0)) * DIMN + hd * DH;
    const bf16_t* bbase = ctxt + (size_t)bh * DH * DH;
    f32x8 acc[2][4] = {};
#pragma unroll
    for (int k0 = 0; k0 < DH; k0 += 32) {
        bf16x16 a0 = load_a_frag(abase + k0, DIMN);
        bf16x16 a1 = load_a_frag(abase + (size_t)16 * DIMN + k0, DIMN);
#pragma unroll
        for (int j = 0; j < 4; ++j) {
            bf16x16 bb = load_bt_frag(bbase + (size_t)(j * 16) * DH + k0, DH);
            acc[0][j] = wmma_bf16(a0, bb, acc[0][j]);
            acc[1][j] = wmma_bf16(a1, bb, acc[1][j]);
        }
    }
    const int hf = lane >> 4, nn = lane & 15;
#pragma unroll
    for (int i = 0; i < 2; ++i)
#pragma unroll
        for (int j = 0; j < 4; ++j)
#pragma unroll
            for (int e = 0; e < 8; ++e) {
                int t = t0 + i * 16 + e + 8 * hf;
                int col = hd * DH + j * 16 + nn;
                attnbf[((size_t)(b * TLEN + t)) * DIMN + col] = (bf16_t)acc[i][j][e];
            }
}

// ---------------- output projection GEMM + bias (async-LDS double-buffered) ----------------
__global__ void __launch_bounds__(256) out_gemm_kernel(const bf16_t* __restrict__ attnbf,
                                                       const bf16_t* __restrict__ wot,
                                                       const float* __restrict__ bo,
                                                       float* __restrict__ out) {
    __shared__ bf16_t tA[2][128 * 64];
    __shared__ bf16_t tB[2][128 * 64];

    const int tid = threadIdx.x;
    const int w = tid >> 5, lane = tid & 31;
    const int wm = w & 3, wn = w >> 2;
    const int m0 = blockIdx.y * 128;
    const int n0 = blockIdx.x * 128;
    const bf16_t* agbl = attnbf + (size_t)m0 * DIMN;
    const bf16_t* bgbl = wot + (size_t)n0 * DIMN;

    f32x8 acc[2][4] = {};

    auto stage = [&](int s) {
        const int b = s & 1;
        const int k0 = s * 64;
#pragma unroll
        for (int i = 0; i < 4; ++i) {
            int chunk = tid + i * 256;
            int row = chunk >> 3;
            int kc  = (chunk & 7) * 8;
            async_load_b128(lds_addr(&tA[b][chunk * 8]),
                            agbl + (size_t)row * DIMN + k0 + kc);
            async_load_b128(lds_addr(&tB[b][chunk * 8]),
                            bgbl + (size_t)row * DIMN + k0 + kc);
        }
    };

    auto compute = [&](int b) {
        const bf16_t* As = &tA[b][(wm * 32) * 64];
        const bf16_t* Bs = &tB[b][(wn * 64) * 64];
#pragma unroll
        for (int kk = 0; kk < 64; kk += 32) {
            bf16x16 a0 = load_a_frag(As + kk, 64);
            bf16x16 a1 = load_a_frag(As + 16 * 64 + kk, 64);
#pragma unroll
            for (int j = 0; j < 4; ++j) {
                bf16x16 bb = load_bt_frag(Bs + (j * 16) * 64 + kk, 64);
                acc[0][j] = wmma_bf16(a0, bb, acc[0][j]);
                acc[1][j] = wmma_bf16(a1, bb, acc[1][j]);
            }
        }
    };

    stage(0);
    wait_async0();
    __syncthreads();
    const int KT = DIMN / 64;
    for (int s = 0; s < KT; ++s) {
        if (s + 1 < KT) stage(s + 1);
        compute(s & 1);
        wait_async0();
        __syncthreads();
    }

    const int hf = lane >> 4, nn = lane & 15;
    const int wm0 = m0 + wm * 32, wn0 = n0 + wn * 64;
#pragma unroll
    for (int i = 0; i < 2; ++i)
#pragma unroll
        for (int j = 0; j < 4; ++j)
#pragma unroll
            for (int e = 0; e < 8; ++e) {
                int r = wm0 + i * 16 + e + 8 * hf;
                int c = wn0 + j * 16 + nn;
                out[(size_t)r * DIMN + c] = acc[i][j][e] + bo[c];
            }
}

// ---------------- host launcher ----------------
extern "C" void kernel_launch(void* const* d_in, const int* in_sizes, int n_in,
                              void* d_out, int out_size, void* d_ws, size_t ws_size,
                              hipStream_t stream) {
    (void)in_sizes; (void)n_in; (void)out_size; (void)ws_size;
    const float* x  = (const float*)d_in[0];
    const float* Wq = (const float*)d_in[1];
    const float* Wk = (const float*)d_in[2];
    const float* Wv = (const float*)d_in[3];
    const float* Wo = (const float*)d_in[4];
    const float* bo = (const float*)d_in[5];
    float* out = (float*)d_out;

    char* ws = (char*)d_ws;
    size_t off = 0;
    auto walloc = [&](size_t bytes) -> char* {
        char* p = ws + off;
        off += (bytes + 255) & ~(size_t)255;
        return p;
    };
    bf16_t* xbf   = (bf16_t*)walloc((size_t)MTOT * DIMN * 2);
    bf16_t* wqkvt = (bf16_t*)walloc((size_t)NQKV * DIMN * 2);
    bf16_t* wot   = (bf16_t*)walloc((size_t)DIMN * DIMN * 2);
    float*  qf    = (float*) walloc((size_t)MTOT * DIMN * 4);
    float*  ktf   = (float*) walloc((size_t)MTOT * DIMN * 4);
    bf16_t* vt    = (bf16_t*)walloc((size_t)MTOT * DIMN * 2);
    bf16_t* qbf   = (bf16_t*)walloc((size_t)MTOT * DIMN * 2);
    bf16_t* kbf   = (bf16_t*)walloc((size_t)MTOT * DIMN * 2);
    bf16_t* ctxt  = (bf16_t*)walloc((size_t)BATCH * HNUM * DH * DH * 2);
    bf16_t* attnb = (bf16_t*)walloc((size_t)MTOT * DIMN * 2);

    cast_f32_bf16_kernel<<<(MTOT * DIMN) / 256, 256, 0, stream>>>(x, xbf, MTOT * DIMN);
    pack_wqkv_kernel<<<(NQKV * DIMN) / 256, 256, 0, stream>>>(Wq, Wk, Wv, wqkvt);
    pack_wo_kernel<<<(DIMN * DIMN) / 256, 256, 0, stream>>>(Wo, wot);
    qkv_gemm_kernel<<<dim3(NQKV / 128, MTOT / 128), 256, 0, stream>>>(xbf, wqkvt, qf, ktf, vt);
    softmax_q_kernel<<<(MTOT * HNUM) / 8, 256, 0, stream>>>(qf, qbf);
    softmax_k_kernel<<<BATCH * HNUM * DH, 256, 0, stream>>>(ktf, kbf);
    context_kernel<<<BATCH * HNUM, 128, 0, stream>>>(kbf, vt, ctxt);
    attn_kernel<<<dim3(TLEN / 128, BATCH * HNUM), 128, 0, stream>>>(qbf, ctxt, attnb);
    out_gemm_kernel<<<dim3(DIMN / 128, MTOT / 128), 256, 0, stream>>>(attnb, wot, bo, out);
}
